// RNN_40089224741305
// MI455X (gfx1250) — matrix operand order repaired and checked by reading, not compiled, not run on hardware
//
#include <hip/hip_runtime.h>
#include <math.h>

// ---------------------------------------------------------------------------
// 2-layer GRU (L=512, B=256, D=H=512) on gfx1250.
// f16 WMMA (V_WMMA_F32_16X16X32_F16) with fp32 accumulation.
// Sequential phase is a 3-stage pipeline across WGPs per 16-row batch tile:
//   A: layer-0 cell (h-side GEMM, x-side from precomputed GI0)
//   C: GI1_t = h0_t @ Wi1
//   B: layer-1 cell (h-side GEMM + GI1)
// Stages hand off through L2-hot f16 ring buffers + release/acquire flags.
// CDNA5 paths: v_wmma, global_load_async_to_lds_b128 (+s_wait_asynccnt),
// global_prefetch_b8.
// ---------------------------------------------------------------------------

typedef __attribute__((ext_vector_type(16))) _Float16 v16h;
typedef __attribute__((ext_vector_type(8)))  _Float16 v8h;
typedef __attribute__((ext_vector_type(8)))  float    v8f;

#define L_STEPS 512
#define BATCH   256
#define DIM     512
#define N3H     1536
#define NTILES  16          // batch tiles
#define RING    4           // ring depth (power of two)
#define PACK_ELEMS (512 * 1536)

// ---- workspace layout (bytes) ---------------------------------------------
#define WS_PACK_OFF   ((size_t)0)
#define WS_PACK_BYTES ((size_t)4 * PACK_ELEMS * 2)                    // 6,291,456
#define WS_H0R_OFF    (WS_PACK_OFF + WS_PACK_BYTES)
#define WS_H0R_BYTES  ((size_t)NTILES * RING * 16 * DIM * 2)          // 1,048,576
#define WS_GI1R_OFF   (WS_H0R_OFF + WS_H0R_BYTES)
#define WS_GI1R_BYTES ((size_t)NTILES * RING * 16 * N3H * 2)          // 3,145,728
#define WS_FLAG_OFF   (WS_GI1R_OFF + WS_GI1R_BYTES)
#define WS_FLAG_COUNT ((size_t)3 * NTILES * L_STEPS)                  // 24,576 u32
#define WS_FLAG_BYTES (WS_FLAG_COUNT * 4)
#define WS_GI0_OFF    (WS_FLAG_OFF + WS_FLAG_BYTES)                   // 10,584,064 (256B aligned)
#define WS_GI0_BYTES  ((size_t)L_STEPS * BATCH * N3H * 2)             // 402,653,184

__device__ __forceinline__ v16h cat16(v8h lo, v8h hi) {
    return __builtin_shufflevector(lo, hi, 0,1,2,3,4,5,6,7,8,9,10,11,12,13,14,15);
}

// A fragment (16x32 f16): M = lane&15; lanes 0-15: K={0..7,16..23},
// lanes 16-31: K={8..15,24..31} -> two contiguous 8-half runs.
__device__ __forceinline__ v16h load_a_lds(const _Float16* __restrict__ buf,
                                           int M, int kbase, int group) {
    const v8h* p0 = (const v8h*)(buf + M * DIM + kbase + group * 8);
    const v8h* p1 = (const v8h*)(buf + M * DIM + kbase + 16 + group * 8);
    return cat16(*p0, *p1);
}

// B fragment (32x16 f16), pre-packed: 16 contiguous halves per lane.
__device__ __forceinline__ v16h load_b_pack(const _Float16* __restrict__ pack,
                                            int tile, int lane) {
    const v8h* p = (const v8h*)(pack + ((size_t)tile * 32 + lane) * 16);
    return cat16(p[0], p[1]);
}

__device__ __forceinline__ v8f wmma_f16(v16h a, v16h b, v8f c) {
    return __builtin_amdgcn_wmma_f32_16x16x32_f16(false, a, false, b,
                                                  (short)0, c, false, false);
}

__device__ __forceinline__ float sigm(float x) { return 1.0f / (1.0f + __expf(-x)); }
__device__ __forceinline__ float tanh_fast(float x) {
    x = fmaxf(fminf(x, 15.0f), -15.0f);
    float e = __expf(-2.0f * x);
    return (1.0f - e) / (1.0f + e);
}

// ---- CDNA5 async global->LDS copy (16B per lane, ASYNCcnt tracked) --------
__device__ __forceinline__ void async_g2l_b128(_Float16* lds_dst, const _Float16* gsrc) {
    unsigned laddr = (unsigned)(size_t)lds_dst;    // flat->LDS: low 32 bits
    asm volatile("global_load_async_to_lds_b128 %0, %1, off"
                 :: "v"(laddr), "v"(gsrc)
                 : "memory");
}
__device__ __forceinline__ void wait_async0() {
    asm volatile("s_wait_asynccnt 0" ::: "memory");
}

// ---- producer/consumer flags ----------------------------------------------
__device__ __forceinline__ void wait_flag(const unsigned* f) {
    if (threadIdx.x == 0) {
        while (__hip_atomic_load(f, __ATOMIC_ACQUIRE, __HIP_MEMORY_SCOPE_AGENT) == 0u)
            __builtin_amdgcn_s_sleep(1);
    }
    __syncthreads();
    __threadfence();        // make partner's stores visible to all lanes' loads
}
__device__ __forceinline__ void set_flag(unsigned* f) {
    __threadfence();        // each thread publishes its stores agent-wide
    __syncthreads();        // (also serves as the LDS read/write barrier)
    if (threadIdx.x == 0)
        __hip_atomic_store(f, 1u, __ATOMIC_RELEASE, __HIP_MEMORY_SCOPE_AGENT);
}

// ---------------------------------------------------------------------------
// GRU cell accumulation for one 16-row tile; returns hnew in registers.
// No barriers, no LDS writes (caller handles state update).
// ---------------------------------------------------------------------------
__device__ __forceinline__ void gru_cell(const _Float16* __restrict__ inbuf,
                                         const _Float16* __restrict__ hbuf,
                                         const _Float16* __restrict__ wiP,
                                         const _Float16* __restrict__ whP,
                                         const float* __restrict__ bh,
                                         const _Float16* __restrict__ gi,
                                         int wave, int lane, float* hnew) {
    const int group = lane >> 4;
    const int nl = lane & 15;
    const int col = wave * 16 + nl;
    v8f ar = {}, az = {}, ain = {}, ahn = {};
    const float br = bh[col], bz = bh[512 + col], bn = bh[1024 + col];
    #pragma unroll
    for (int g = 0; g < 8; ++g) { ar[g] = br; az[g] = bz; ahn[g] = bn; }

    if (gi) {   // x-side GEMM precomputed ([16][1536] f16, no bias)
        #pragma unroll
        for (int g = 0; g < 8; ++g) {
            const _Float16* p = gi + (size_t)(g + group * 8) * N3H;
            ar[g]  += (float)p[col];
            az[g]  += (float)p[512 + col];
            ain[g] += (float)p[1024 + col];
        }
    } else {    // x-side K loop (K = 512) from LDS
        for (int kt = 0; kt < 16; ++kt) {
            v16h a   = load_a_lds(inbuf, nl, kt * 32, group);
            v16h br_ = load_b_pack(wiP, kt * 96 + wave,      lane);
            v16h bz_ = load_b_pack(wiP, kt * 96 + 32 + wave, lane);
            v16h bn_ = load_b_pack(wiP, kt * 96 + 64 + wave, lane);
            ar  = wmma_f16(a, br_, ar);
            az  = wmma_f16(a, bz_, az);
            ain = wmma_f16(a, bn_, ain);
        }
    }
    for (int kt = 0; kt < 16; ++kt) {           // hidden-side K loop
        v16h a   = load_a_lds(hbuf, nl, kt * 32, group);
        v16h br_ = load_b_pack(whP, kt * 96 + wave,      lane);
        v16h bz_ = load_b_pack(whP, kt * 96 + 32 + wave, lane);
        v16h bn_ = load_b_pack(whP, kt * 96 + 64 + wave, lane);
        ar  = wmma_f16(a, br_, ar);
        az  = wmma_f16(a, bz_, az);
        ahn = wmma_f16(a, bn_, ahn);
    }
    #pragma unroll
    for (int g = 0; g < 8; ++g) {               // gates
        int Mg = g + group * 8;
        float r = sigm(ar[g]);
        float z = sigm(az[g]);
        float n = tanh_fast(ain[g] + r * ahn[g]);
        float hp = (float)hbuf[Mg * DIM + col];
        hnew[g] = (1.0f - z) * n + z * hp;
    }
}

// ---------------------------------------------------------------------------
// Kernel: pack fp32 weights [512x1536] into f16 WMMA B-fragment order.
// ---------------------------------------------------------------------------
__global__ void pack_weights(const float* __restrict__ Wi0,
                             const float* __restrict__ Wh0,
                             const float* __restrict__ Wi1,
                             const float* __restrict__ Wh1,
                             _Float16* __restrict__ wpack) {
    size_t idx = (size_t)blockIdx.x * blockDim.x + threadIdx.x;
    if (idx >= (size_t)4 * PACK_ELEMS) return;
    int m = (int)(idx / PACK_ELEMS);
    size_t r = idx % PACK_ELEMS;
    int tile = (int)(r >> 9);
    int lane = (int)((r >> 4) & 31);
    int h    = (int)(r & 15);
    int kt = tile / 96, nt = tile % 96;
    int K = kt * 32 + (h & 7) + ((h >> 3) << 4) + (lane >> 4) * 8;
    int N = nt * 16 + (lane & 15);
    const float* src = (m == 0) ? Wi0 : (m == 1) ? Wh0 : (m == 2) ? Wi1 : Wh1;
    wpack[idx] = (_Float16)src[(size_t)K * N3H + N];
}

__global__ void zero_flags(unsigned* __restrict__ flags) {
    size_t i = (size_t)blockIdx.x * blockDim.x + threadIdx.x;
    if (i < WS_FLAG_COUNT) flags[i] = 0u;
}

// ---------------------------------------------------------------------------
// Kernel: GI0 = X @ Wi0  (fully parallel, one wave per 16x16 tile).
// ---------------------------------------------------------------------------
__global__ __launch_bounds__(256) void gi0_gemm(const float* __restrict__ x,
                                                const _Float16* __restrict__ wi0p,
                                                _Float16* __restrict__ gi0) {
    int waveGlobal = blockIdx.x * 8 + (threadIdx.x >> 5);
    int lane  = threadIdx.x & 31;
    int mt = waveGlobal / 96;
    int nt = waveGlobal % 96;
    int group = lane >> 4, nl = lane & 15;
    const float* xr = x + ((size_t)mt * 16 + nl) * DIM;
    v8f acc = {};
    for (int kt = 0; kt < 16; ++kt) {
        int kb = kt * 32 + group * 8;
        v16h a;
        #pragma unroll
        for (int j = 0; j < 8; ++j) a[j]     = (_Float16)xr[kb + j];
        #pragma unroll
        for (int j = 0; j < 8; ++j) a[8 + j] = (_Float16)xr[kb + 16 + j];
        acc = wmma_f16(a, load_b_pack(wi0p, kt * 96 + nt, lane), acc);
    }
    #pragma unroll
    for (int g = 0; g < 8; ++g) {
        size_t Mg = (size_t)mt * 16 + g + group * 8;
        gi0[Mg * N3H + nt * 16 + nl] = (_Float16)acc[g];
    }
}

// ---------------------------------------------------------------------------
// Pipelined recurrence: 48 blocks x 1024 threads. block = tile*3 + role.
//   role 0: layer-0 cell    -> h0 ring   (+flag fA)
//   role 1: GI1 = h0 @ Wi1  -> gi1 ring  (+flag fC)  [consumes fA, backpress fB]
//   role 2: layer-1 cell    -> y, h1     (+flag fB)  [consumes fC]
// ---------------------------------------------------------------------------
__global__ __launch_bounds__(1024) void gru_pipe(const float* __restrict__ carry_in,
                                                 const float* __restrict__ x,
                                                 const float* __restrict__ bh0,
                                                 const float* __restrict__ bh1,
                                                 const _Float16* __restrict__ wpack,
                                                 const _Float16* __restrict__ gi0,
                                                 int use_gi0,
                                                 _Float16* __restrict__ h0ring,
                                                 _Float16* __restrict__ gi1ring,
                                                 unsigned* __restrict__ flags,
                                                 float* __restrict__ out) {
    __shared__ __align__(16) _Float16 s0[16 * DIM];   // state / staging
    __shared__ __align__(16) _Float16 s1[16 * DIM];   // x staging (role 0, no-gi0)

    const int tid  = threadIdx.x;
    const int wave = tid >> 5;
    const int lane = tid & 31;
    const int tile = blockIdx.x / 3;
    const int role = blockIdx.x % 3;
    const int m0   = tile * 16;
    const int group = lane >> 4, nl = lane & 15;
    const int col = wave * 16 + nl;

    const _Float16* Wi0p = wpack;
    const _Float16* Wh0p = wpack + 1 * PACK_ELEMS;
    const _Float16* Wi1p = wpack + 2 * PACK_ELEMS;
    const _Float16* Wh1p = wpack + 3 * PACK_ELEMS;

    unsigned* fA = flags;
    unsigned* fC = flags + NTILES * L_STEPS;
    unsigned* fB = flags + 2 * NTILES * L_STEPS;

    float* yout = out + (size_t)BATCH * 1024;
    float hnew[8];

    if (role == 0) {  // ---------------- layer-0 cell ----------------
        for (int i = tid; i < 16 * DIM; i += 1024) {
            int rr = i >> 9, cc = i & 511;
            s0[i] = (_Float16)carry_in[(size_t)(m0 + rr) * 1024 + cc];
        }
        __syncthreads();
        for (int t = 0; t < L_STEPS; ++t) {
            if (t >= RING) wait_flag(&fC[tile * L_STEPS + t - RING]);  // ring free?
            const _Float16* gi = nullptr;
            if (use_gi0) {
                gi = gi0 + ((size_t)t * BATCH + m0) * N3H;
                if (t + 1 < L_STEPS) {   // prefetch next step's 48KB GI0 slice
                    const char* pn = (const char*)(gi0 + ((size_t)(t + 1) * BATCH + m0) * N3H);
                    __builtin_prefetch(pn + (size_t)tid * 48, 0, 3);
                }
            } else {
                for (int i = tid; i < 16 * DIM; i += 1024) {
                    int rr = i >> 9, cc = i & 511;
                    s1[i] = (_Float16)x[((size_t)t * BATCH + m0 + rr) * DIM + cc];
                }
                __syncthreads();
            }
            gru_cell(s1, s0, Wi0p, Wh0p, bh0, gi, wave, lane, hnew);
            __syncthreads();                    // all reads of s0 done
            _Float16* ring = h0ring + ((size_t)tile * RING + (t & (RING - 1))) * 16 * DIM;
            #pragma unroll
            for (int g = 0; g < 8; ++g) {
                int Mg = g + group * 8;
                _Float16 hv = (_Float16)hnew[g];
                s0[Mg * DIM + col]   = hv;
                ring[Mg * DIM + col] = hv;
            }
            set_flag(&fA[tile * L_STEPS + t]);
        }
        __syncthreads();
        for (int i = tid; i < 16 * DIM; i += 1024) {
            int rr = i >> 9, cc = i & 511;
            out[(size_t)(m0 + rr) * 1024 + cc] = (float)s0[i];
        }
    } else if (role == 1) {  // ---------------- GI1 = h0_t @ Wi1 ----------------
        for (int t = 0; t < L_STEPS; ++t) {
            if (t >= RING) wait_flag(&fB[tile * L_STEPS + t - RING]);  // gi1 ring free?
            wait_flag(&fA[tile * L_STEPS + t]);                        // h0_t ready
            const _Float16* hr = h0ring + ((size_t)tile * RING + (t & (RING - 1))) * 16 * DIM;
            // async global->LDS stage of the 16KB h0 tile (ASYNCcnt path)
            async_g2l_b128(s0 + (size_t)tid * 8, hr + (size_t)tid * 8);
            wait_async0();
            __syncthreads();
            v8f a0 = {}, a1 = {}, a2 = {};
            for (int kt = 0; kt < 16; ++kt) {
                v16h a = load_a_lds(s0, nl, kt * 32, group);
                a0 = wmma_f16(a, load_b_pack(Wi1p, kt * 96 + wave,      lane), a0);
                a1 = wmma_f16(a, load_b_pack(Wi1p, kt * 96 + 32 + wave, lane), a1);
                a2 = wmma_f16(a, load_b_pack(Wi1p, kt * 96 + 64 + wave, lane), a2);
            }
            _Float16* gr = gi1ring + ((size_t)tile * RING + (t & (RING - 1))) * 16 * N3H;
            #pragma unroll
            for (int g = 0; g < 8; ++g) {
                int Mg = g + group * 8;
                gr[(size_t)Mg * N3H + col]        = (_Float16)a0[g];
                gr[(size_t)Mg * N3H + 512 + col]  = (_Float16)a1[g];
                gr[(size_t)Mg * N3H + 1024 + col] = (_Float16)a2[g];
            }
            set_flag(&fC[tile * L_STEPS + t]);
        }
    } else {  // ---------------- layer-1 cell ----------------
        for (int i = tid; i < 16 * DIM; i += 1024) {
            int rr = i >> 9, cc = i & 511;
            s0[i] = (_Float16)carry_in[(size_t)(m0 + rr) * 1024 + 512 + cc];
        }
        __syncthreads();
        for (int t = 0; t < L_STEPS; ++t) {
            wait_flag(&fC[tile * L_STEPS + t]);                        // gi1_t ready
            const _Float16* gi = gi1ring + ((size_t)tile * RING + (t & (RING - 1))) * 16 * N3H;
            gru_cell(nullptr, s0, nullptr, Wh1p, bh1, gi, wave, lane, hnew);
            __syncthreads();                    // all reads of s0 done
            #pragma unroll
            for (int g = 0; g < 8; ++g) {
                int Mg = g + group * 8;
                s0[Mg * DIM + col] = (_Float16)hnew[g];
                yout[((size_t)t * BATCH + m0 + Mg) * DIM + col] = hnew[g];
            }
            set_flag(&fB[tile * L_STEPS + t]);
        }
        __syncthreads();
        for (int i = tid; i < 16 * DIM; i += 1024) {
            int rr = i >> 9, cc = i & 511;
            out[(size_t)(m0 + rr) * 1024 + 512 + cc] = (float)s0[i];
        }
    }
}

// ---------------------------------------------------------------------------
// Fallback fused recurrence (16 blocks x 1024 threads), no gi0.
// ---------------------------------------------------------------------------
__global__ __launch_bounds__(1024) void gru_seq(const float* __restrict__ carry_in,
                                                const float* __restrict__ x,
                                                const float* __restrict__ bh0,
                                                const float* __restrict__ bh1,
                                                const _Float16* __restrict__ wpack,
                                                float* __restrict__ out) {
    __shared__ __align__(16) _Float16 inbuf[16 * DIM];
    __shared__ __align__(16) _Float16 h0lds[16 * DIM];
    __shared__ __align__(16) _Float16 h1lds[16 * DIM];

    const int tid  = threadIdx.x;
    const int wave = tid >> 5;
    const int lane = tid & 31;
    const int m0   = blockIdx.x * 16;
    const int group = lane >> 4, nl = lane & 15;
    const int col = wave * 16 + nl;

    const _Float16* Wi0p = wpack;
    const _Float16* Wh0p = wpack + 1 * PACK_ELEMS;
    const _Float16* Wi1p = wpack + 2 * PACK_ELEMS;
    const _Float16* Wh1p = wpack + 3 * PACK_ELEMS;

    for (int i = tid; i < 16 * DIM; i += 1024) {
        int rr = i >> 9, cc = i & 511;
        h0lds[i] = (_Float16)carry_in[(size_t)(m0 + rr) * 1024 + cc];
        h1lds[i] = (_Float16)carry_in[(size_t)(m0 + rr) * 1024 + 512 + cc];
    }
    __syncthreads();

    float* yout = out + (size_t)BATCH * 1024;
    float hnew[8];

    for (int t = 0; t < L_STEPS; ++t) {
        for (int i = tid; i < 16 * DIM; i += 1024) {
            int rr = i >> 9, cc = i & 511;
            inbuf[i] = (_Float16)x[((size_t)t * BATCH + m0 + rr) * DIM + cc];
        }
        __syncthreads();
        gru_cell(inbuf, h0lds, Wi0p, Wh0p, bh0, nullptr, wave, lane, hnew);
        __syncthreads();
        #pragma unroll
        for (int g = 0; g < 8; ++g) {
            int Mg = g + group * 8;
            _Float16 hv = (_Float16)hnew[g];
            h0lds[Mg * DIM + col] = hv;
            inbuf[Mg * DIM + col] = hv;     // layer-1 input
        }
        __syncthreads();
        gru_cell(inbuf, h1lds, Wi1p, Wh1p, bh1, nullptr, wave, lane, hnew);
        __syncthreads();
        #pragma unroll
        for (int g = 0; g < 8; ++g) {
            int Mg = g + group * 8;
            h1lds[Mg * DIM + col] = (_Float16)hnew[g];
            yout[((size_t)t * BATCH + m0 + Mg) * DIM + col] = hnew[g];
        }
        __syncthreads();
    }

    for (int i = tid; i < 16 * DIM; i += 1024) {
        int rr = i >> 9, cc = i & 511;
        out[(size_t)(m0 + rr) * 1024 + cc]       = (float)h0lds[i];
        out[(size_t)(m0 + rr) * 1024 + 512 + cc] = (float)h1lds[i];
    }
}

// ---------------------------------------------------------------------------
extern "C" void kernel_launch(void* const* d_in, const int* in_sizes, int n_in,
                              void* d_out, int out_size, void* d_ws, size_t ws_size,
                              hipStream_t stream) {
    const float* carry = (const float*)d_in[0];
    const float* x     = (const float*)d_in[1];
    const float* Wi0   = (const float*)d_in[2];
    const float* Wh0   = (const float*)d_in[3];
    const float* bh0   = (const float*)d_in[4];
    const float* Wi1   = (const float*)d_in[5];
    const float* Wh1   = (const float*)d_in[6];
    const float* bh1   = (const float*)d_in[7];
    float* out = (float*)d_out;

    char* ws = (char*)d_ws;
    _Float16* wpack   = (_Float16*)(ws + WS_PACK_OFF);
    _Float16* h0ring  = (_Float16*)(ws + WS_H0R_OFF);
    _Float16* gi1ring = (_Float16*)(ws + WS_GI1R_OFF);
    unsigned* flags   = (unsigned*)(ws + WS_FLAG_OFF);
    _Float16* gi0     = (_Float16*)(ws + WS_GI0_OFF);

    const int pipe_ok = (ws_size >= WS_GI0_OFF) ? 1 : 0;
    const int use_gi0 = (ws_size >= WS_GI0_OFF + WS_GI0_BYTES) ? 1 : 0;

    pack_weights<<<(4 * PACK_ELEMS + 255) / 256, 256, 0, stream>>>(
        Wi0, Wh0, Wi1, Wh1, wpack);

    if (pipe_ok) {
        zero_flags<<<(int)((WS_FLAG_COUNT + 255) / 256), 256, 0, stream>>>(flags);
        if (use_gi0)
            gi0_gemm<<<98304, 256, 0, stream>>>(x, wpack, gi0);   // 786432 tiles / 8 waves
        gru_pipe<<<3 * NTILES, 1024, 0, stream>>>(carry, x, bh0, bh1, wpack,
                                                  gi0, use_gi0, h0ring, gi1ring,
                                                  flags, out);
    } else {
        gru_seq<<<NTILES, 1024, 0, stream>>>(carry, x, bh0, bh1, wpack, out);
    }
}
